// Net_8435315769443
// MI455X (gfx1250) — matrix operand order repaired and checked by reading, not compiled, not run on hardware
//
#include <hip/hip_runtime.h>

typedef __attribute__((ext_vector_type(16))) _Float16 v16h;
typedef __attribute__((ext_vector_type(8)))  _Float16 v8h;
typedef __attribute__((ext_vector_type(8)))  float    v8f;

#define N_NODES 100000
#define IN_C    512
#define HID     256
#define OUT_C   64
#define N_EDGES 1600000
#define K_STEPS 10
#define ALPHA   0.1f

// async global->LDS copy of 16 bytes (per-lane), tracked by ASYNCcnt
#define ASYNC_B128(ldsOff, gOff, sBase)                                   \
  asm volatile("global_load_async_to_lds_b128 %0, %1, %2"                 \
               :: "v"(ldsOff), "v"(gOff), "s"(sBase) : "memory")
#define WAIT_ASYNC() asm volatile("s_wait_asynccnt 0" ::: "memory")

// ---------------- fp32 -> f16 weight conversion ----------------
__global__ void cvt_f32_f16(const float* __restrict__ s, _Float16* __restrict__ d, int n) {
  int i = blockIdx.x * blockDim.x + threadIdx.x;
  if (i < n) d[i] = (_Float16)s[i];
}

// ---------------- GEMM1: h = relu(x @ W1^T + b1), f16 output ----------------
// grid: ceil(N/128) blocks of 256 threads (8 waves). Wave w owns rows [w*16, w*16+16).
// Double-buffered LDS; W1 tiles via async global->LDS, x tiles via VGPR (needs cvt).
// B fragments software-pipelined one WMMA ahead so ds_load latency hides under WMMA.
__global__ __launch_bounds__(256)
void gemm1_relu(const float* __restrict__ x, const _Float16* __restrict__ w1h,
                const float* __restrict__ b1, _Float16* __restrict__ h16) {
  __shared__ __attribute__((aligned(16))) _Float16 xs[2][128 * 32];   // 2x 8 KB
  __shared__ __attribute__((aligned(16))) _Float16 ws[2][HID * 32];   // 2x 16 KB
  const int tid   = threadIdx.x;
  const int wave  = tid >> 5;
  const int lane  = tid & 31;
  const int lhalf = lane & 15;
  const int lhi   = lane >> 4;            // 0: lanes 0-15, 1: lanes 16-31
  const int mBlock = blockIdx.x * 128;
  const int NCHUNK = IN_C / 32;           // 16

  v8f acc[16];
  #pragma unroll
  for (int t = 0; t < 16; ++t) acc[t] = (v8f){};

  float2 xreg[8];

  auto issue_w = [&](int k0, int b) {
    #pragma unroll
    for (int i = 0; i < 4; ++i) {
      int idx = tid + i * 256;            // 1024 x b128 per tile
      int n = idx >> 2, u = idx & 3;      // row n, 16B chunk u
      unsigned ldsOff = (unsigned)(uintptr_t)&ws[b][n * 32 + u * 8];
      unsigned gOff   = (unsigned)((n * IN_C + k0) * 2 + u * 16);
      ASYNC_B128(ldsOff, gOff, w1h);
    }
  };
  auto issue_x = [&](int k0) {
    #pragma unroll
    for (int i = 0; i < 8; ++i) {
      int idx = tid + i * 256;            // pair index, 0..2047
      int r = idx >> 4, kp = idx & 15;
      int row = mBlock + r; if (row >= N_NODES) row = N_NODES - 1;
      xreg[i] = *(const float2*)(x + (size_t)row * IN_C + k0 + 2 * kp);
    }
  };
  auto store_x = [&](int b) {
    #pragma unroll
    for (int i = 0; i < 8; ++i) {
      int idx = tid + i * 256;
      union { _Float16 h[2]; unsigned u; } pk;
      pk.h[0] = (_Float16)xreg[i].x;
      pk.h[1] = (_Float16)xreg[i].y;
      ((unsigned*)xs[b])[idx] = pk.u;
    }
  };

  // prologue: stage chunk 0 into buffer 0
  issue_w(0, 0);
  issue_x(0);
  store_x(0);
  WAIT_ASYNC();
  __syncthreads();

  int buf = 0;
  for (int c = 0; c < NCHUNK; ++c) {
    const bool more = (c + 1 < NCHUNK);
    if (more) {
      issue_w((c + 1) * 32, buf ^ 1);     // async DMA straight into LDS
      issue_x((c + 1) * 32);              // global loads into VGPRs
    }

    // compute on current buffer while next chunk is in flight
    const int arow = wave * 16 + lhalf;
    v8h alo = *(const v8h*)&xs[buf][arow * 32 + lhi * 16];
    v8h ahi = *(const v8h*)&xs[buf][arow * 32 + lhi * 16 + 8];
    v16h afrag = __builtin_shufflevector(alo, ahi, 0,1,2,3,4,5,6,7,8,9,10,11,12,13,14,15);

    // B pipeline: fragment t+1 loads issue before WMMA t
    v8h blo = *(const v8h*)&ws[buf][lhalf * 32 + lhi * 16];
    v8h bhi = *(const v8h*)&ws[buf][lhalf * 32 + lhi * 16 + 8];
    #pragma unroll
    for (int t = 0; t < 16; ++t) {
      v16h bfrag = __builtin_shufflevector(blo, bhi, 0,1,2,3,4,5,6,7,8,9,10,11,12,13,14,15);
      if (t + 1 < 16) {
        const int n = (t + 1) * 16 + lhalf;
        blo = *(const v8h*)&ws[buf][n * 32 + lhi * 16];
        bhi = *(const v8h*)&ws[buf][n * 32 + lhi * 16 + 8];
      }
      acc[t] = __builtin_amdgcn_wmma_f32_16x16x32_f16(
          false, afrag, false, bfrag, (short)0, acc[t], false, false);
    }

    if (more) store_x(buf ^ 1);           // cvt + ds_store after WMMAs issued
    WAIT_ASYNC();
    __syncthreads();
    buf ^= 1;
  }

  // epilogue: bias + relu + cvt f16, coalesced row-contiguous stores
  #pragma unroll
  for (int t = 0; t < 16; ++t) {
    const int n = t * 16 + lhalf;
    const float bias = b1[n];
    #pragma unroll
    for (int v = 0; v < 8; ++v) {
      int row = mBlock + wave * 16 + v + lhi * 8;
      if (row < N_NODES) {
        float val = acc[t][v] + bias;
        val = val > 0.f ? val : 0.f;
        h16[(size_t)row * HID + n] = (_Float16)val;
      }
    }
  }
}

// ---------------- GEMM2: z = h @ W2^T + b2, fp32 output ----------------
// Both operands already f16 in global: all staging via async global->LDS DMA.
__global__ __launch_bounds__(256)
void gemm2(const _Float16* __restrict__ h16, const _Float16* __restrict__ w2h,
           const float* __restrict__ b2, float* __restrict__ z) {
  __shared__ __attribute__((aligned(16))) _Float16 hs[2][128 * 32];    // 2x 8 KB
  __shared__ __attribute__((aligned(16))) _Float16 w2s[2][OUT_C * 32]; // 2x 4 KB
  const int tid   = threadIdx.x;
  const int wave  = tid >> 5;
  const int lane  = tid & 31;
  const int lhalf = lane & 15;
  const int lhi   = lane >> 4;
  const int mBlock = blockIdx.x * 128;
  const int NCHUNK = HID / 32;            // 8

  v8f acc[4];
  #pragma unroll
  for (int t = 0; t < 4; ++t) acc[t] = (v8f){};

  auto issue_h = [&](int k0, int b) {
    #pragma unroll
    for (int i = 0; i < 2; ++i) {
      int idx = tid + i * 256;            // 512 x b128 per tile
      int r = idx >> 2, u = idx & 3;
      int row = mBlock + r; if (row >= N_NODES) row = N_NODES - 1;
      unsigned ldsOff = (unsigned)(uintptr_t)&hs[b][r * 32 + u * 8];
      unsigned gOff   = (unsigned)(((size_t)row * HID + k0) * 2 + u * 16);
      ASYNC_B128(ldsOff, gOff, h16);
    }
  };
  auto issue_w2 = [&](int k0, int b) {
    int idx = tid;                         // 256 x b128 per tile
    int n = idx >> 2, u = idx & 3;
    unsigned ldsOff = (unsigned)(uintptr_t)&w2s[b][n * 32 + u * 8];
    unsigned gOff   = (unsigned)((n * HID + k0) * 2 + u * 16);
    ASYNC_B128(ldsOff, gOff, w2h);
  };

  issue_h(0, 0);
  issue_w2(0, 0);
  WAIT_ASYNC();
  __syncthreads();

  int buf = 0;
  for (int c = 0; c < NCHUNK; ++c) {
    const bool more = (c + 1 < NCHUNK);
    if (more) {
      issue_h((c + 1) * 32, buf ^ 1);
      issue_w2((c + 1) * 32, buf ^ 1);
    }

    const int arow = wave * 16 + lhalf;
    v8h alo = *(const v8h*)&hs[buf][arow * 32 + lhi * 16];
    v8h ahi = *(const v8h*)&hs[buf][arow * 32 + lhi * 16 + 8];
    v16h afrag = __builtin_shufflevector(alo, ahi, 0,1,2,3,4,5,6,7,8,9,10,11,12,13,14,15);

    v8h blo = *(const v8h*)&w2s[buf][lhalf * 32 + lhi * 16];
    v8h bhi = *(const v8h*)&w2s[buf][lhalf * 32 + lhi * 16 + 8];
    #pragma unroll
    for (int t = 0; t < 4; ++t) {
      v16h bfrag = __builtin_shufflevector(blo, bhi, 0,1,2,3,4,5,6,7,8,9,10,11,12,13,14,15);
      if (t + 1 < 4) {
        const int n = (t + 1) * 16 + lhalf;
        blo = *(const v8h*)&w2s[buf][n * 32 + lhi * 16];
        bhi = *(const v8h*)&w2s[buf][n * 32 + lhi * 16 + 8];
      }
      acc[t] = __builtin_amdgcn_wmma_f32_16x16x32_f16(
          false, afrag, false, bfrag, (short)0, acc[t], false, false);
    }

    WAIT_ASYNC();
    __syncthreads();
    buf ^= 1;
  }

  #pragma unroll
  for (int t = 0; t < 4; ++t) {
    const int n = t * 16 + lhalf;
    const float bias = b2[n];
    #pragma unroll
    for (int v = 0; v < 8; ++v) {
      int row = mBlock + wave * 16 + v + lhi * 8;
      if (row < N_NODES)
        z[(size_t)row * OUT_C + n] = acc[t][v] + bias;
    }
  }
}

// ---------------- APPNP propagation ----------------
__global__ void deg_init(float* __restrict__ deg) {
  int i = blockIdx.x * blockDim.x + threadIdx.x;
  if (i < N_NODES) deg[i] = 1.0f;            // self-loop contribution
}
__global__ void deg_edges(const int* __restrict__ dst, float* __restrict__ deg) {
  int e = blockIdx.x * blockDim.x + threadIdx.x;
  if (e < N_EDGES) atomicAdd(&deg[dst[e]], 1.0f);
}
__global__ void make_dinv(float* __restrict__ deg) {
  int i = blockIdx.x * blockDim.x + threadIdx.x;
  if (i < N_NODES) deg[i] = rsqrtf(deg[i]);  // deg >= 1 always
}
__global__ void edge_w(const int* __restrict__ src, const int* __restrict__ dst,
                       const float* __restrict__ dinv, float* __restrict__ w) {
  int e = blockIdx.x * blockDim.x + threadIdx.x;
  if (e < N_EDGES) w[e] = dinv[src[e]] * dinv[dst[e]];
}
// hnew = alpha*z + (1-alpha) * selfweight * hcur  (initializes hnew fully)
__global__ void prop_self(const float* __restrict__ z, const float* __restrict__ hcur,
                          const float* __restrict__ dinv, float* __restrict__ hnew) {
  int idx = blockIdx.x * blockDim.x + threadIdx.x;
  if (idx < N_NODES * OUT_C) {
    int i = idx >> 6;
    float di = dinv[i];
    hnew[idx] = ALPHA * z[idx] + (1.0f - ALPHA) * di * di * hcur[idx];
  }
}
// hnew[dst] += (1-alpha) * w[e] * hcur[src], one thread per (edge, channel)
__global__ void prop_edges(const int* __restrict__ src, const int* __restrict__ dst,
                           const float* __restrict__ w, const float* __restrict__ hcur,
                           float* __restrict__ hnew) {
  int idx = blockIdx.x * blockDim.x + threadIdx.x;
  if (idx < N_EDGES * OUT_C) {
    int e = idx >> 6;
    int c = idx & 63;
    int s = src[e], d = dst[e];
    float val = (1.0f - ALPHA) * w[e] * hcur[s * OUT_C + c];
    atomicAdd(&hnew[d * OUT_C + c], val);
  }
}

// ---------------- launch ----------------
static inline char* align256(char* p) {
  return (char*)(((uintptr_t)p + 255) & ~(uintptr_t)255);
}

extern "C" void kernel_launch(void* const* d_in, const int* in_sizes, int n_in,
                              void* d_out, int out_size, void* d_ws, size_t ws_size,
                              hipStream_t stream) {
  const float* x    = (const float*)d_in[0];
  const int*   eidx = (const int*)d_in[1];           // [2, E]
  const float* W1   = (const float*)d_in[2];
  const float* b1   = (const float*)d_in[3];
  const float* W2   = (const float*)d_in[4];
  const float* b2   = (const float*)d_in[5];
  const int* src = eidx;
  const int* dst = eidx + N_EDGES;

  // workspace carve-up
  char* p = (char*)d_ws;
  _Float16* w1h = (_Float16*)p;  p = align256(p + (size_t)HID * IN_C * 2);
  _Float16* w2h = (_Float16*)p;  p = align256(p + (size_t)OUT_C * HID * 2);
  _Float16* h16 = (_Float16*)p;  p = align256(p + (size_t)N_NODES * HID * 2);
  float*    z   = (float*)p;     p = align256(p + (size_t)N_NODES * OUT_C * 4);
  float*    dinv= (float*)p;     p = align256(p + (size_t)N_NODES * 4);
  float*    ew  = (float*)p;     p = align256(p + (size_t)N_EDGES * 4);
  float*    hA  = (float*)p;     p = align256(p + (size_t)N_NODES * OUT_C * 4);
  float*    hB  = (float*)p;     p = align256(p + (size_t)N_NODES * OUT_C * 4);

  // weights -> f16
  cvt_f32_f16<<<(HID * IN_C + 255) / 256, 256, 0, stream>>>(W1, w1h, HID * IN_C);
  cvt_f32_f16<<<(OUT_C * HID + 255) / 256, 256, 0, stream>>>(W2, w2h, OUT_C * HID);

  // MLP via WMMA
  const int gemmGrid = (N_NODES + 127) / 128;
  gemm1_relu<<<gemmGrid, 256, 0, stream>>>(x, w1h, b1, h16);
  gemm2<<<gemmGrid, 256, 0, stream>>>(h16, w2h, b2, z);

  // GCN normalization
  deg_init <<<(N_NODES + 255) / 256, 256, 0, stream>>>(dinv);
  deg_edges<<<(N_EDGES + 255) / 256, 256, 0, stream>>>(dst, dinv);
  make_dinv<<<(N_NODES + 255) / 256, 256, 0, stream>>>(dinv);
  edge_w   <<<(N_EDGES + 255) / 256, 256, 0, stream>>>(src, dst, dinv, ew);

  // K power-iteration steps; last one writes d_out
  const int selfGrid = (N_NODES * OUT_C + 255) / 256;
  const int edgeGrid = (N_EDGES * OUT_C + 255) / 256;
  const float* cur = z;
  for (int it = 0; it < K_STEPS; ++it) {
    float* nxt = (it == K_STEPS - 1) ? (float*)d_out : ((it & 1) ? hB : hA);
    prop_self <<<selfGrid, 256, 0, stream>>>(z, cur, dinv, nxt);
    prop_edges<<<edgeGrid, 256, 0, stream>>>(src, dst, ew, cur, nxt);
    cur = nxt;
  }
}
